// SelfAttentionLayer_46523085750281
// MI455X (gfx1250) — matrix operand order repaired and checked by reading, not compiled, not run on hardware
//
#include <hip/hip_runtime.h>
#include <hip/hip_bf16.h>

// ---------------------------------------------------------------------------
// Self-attention layer (pre-LN, causal, need_weights) for MI455X / gfx1250.
// All matmuls via v_wmma_f32_16x16x32_f16 (wave32 WMMA, f32 accumulate).
// B=4, S=1024, D=1024, H=16, HD=64.
// ---------------------------------------------------------------------------

static constexpr int B_  = 4;
static constexpr int S_  = 1024;
static constexpr int D_  = 1024;
static constexpr int H_  = 16;
static constexpr int HD_ = 64;
static constexpr float LN_EPS_ = 1e-5f;

typedef __attribute__((ext_vector_type(16))) _Float16 v16h;
typedef __attribute__((ext_vector_type(8)))  _Float16 v8h;
typedef __attribute__((ext_vector_type(8)))  float    v8f;

// ---- WMMA wrapper: D = A(16x32 f16) * B(32x16 f16) + C(16x16 f32) ----------
__device__ __forceinline__ v8f wmma_f16(v16h a, v16h b, v8f c) {
    return __builtin_amdgcn_wmma_f32_16x16x32_f16(
        /*neg_a=*/false, a, /*neg_b=*/false, b,
        /*c_mod=*/(short)0, c, /*reuse_a=*/false, /*reuse_b=*/false);
}

// ---- A fragment (16x32, M x K), source row-major M x K, ld = ldk -----------
// ISA layout: lane r=lane%16 holds row M=r; sel=lane/16 selects K chunks
// {sel*8 .. sel*8+7} and {16+sel*8 .. 16+sel*8+7}.
__device__ __forceinline__ v16h load_a_frag(const _Float16* tile, int ldk) {
    int lane = threadIdx.x & 31;
    int r = lane & 15, sel = lane >> 4;
    const _Float16* p = tile + (size_t)r * ldk + sel * 8;
    v8h lo = *(const v8h*)(p);
    v8h hi = *(const v8h*)(p + 16);
    v16h f;
#pragma unroll
    for (int j = 0; j < 8; ++j) { f[j] = lo[j]; f[j + 8] = hi[j]; }
    return f;
}

// ---- B fragment (32x16, K x N), source row-major N x K ("B^T trick") -------
// ISA layout: lane r=lane%16 holds column N=r; sel=lane/16 selects K chunks
// {sel*8 .. sel*8+7} and {16+sel*8 .. 16+sel*8+7}.
__device__ __forceinline__ v16h load_b_frag(const _Float16* tile, int ldk) {
    int lane = threadIdx.x & 31;
    int r = lane & 15, sel = lane >> 4;
    const _Float16* p = tile + (size_t)r * ldk + sel * 8;
    v8h lo = *(const v8h*)(p);
    v8h hi = *(const v8h*)(p + 16);
    v16h f;
#pragma unroll
    for (int j = 0; j < 8; ++j) { f[j] = lo[j]; f[j + 8] = hi[j]; }
    return f;
}

// ---------------------------------------------------------------------------
// Kernel 1: LayerNorm. One block per (b,s) row. Output xn in f16.
// ---------------------------------------------------------------------------
__global__ void ln_kernel(const float* __restrict__ x,
                          const float* __restrict__ g,
                          const float* __restrict__ bt,
                          _Float16* __restrict__ xn) {
    const size_t row = blockIdx.x;
    const float* xr = x + row * D_;
    float lsum = 0.f, lsq = 0.f;
    for (int i = threadIdx.x; i < D_; i += 256) {
        float v = xr[i];
        lsum += v; lsq += v * v;
    }
#pragma unroll
    for (int m = 16; m >= 1; m >>= 1) {
        lsum += __shfl_xor(lsum, m, 32);
        lsq  += __shfl_xor(lsq,  m, 32);
    }
    __shared__ float s1[8], s2[8];
    int wave = threadIdx.x >> 5, lane = threadIdx.x & 31;
    if (lane == 0) { s1[wave] = lsum; s2[wave] = lsq; }
    __syncthreads();
    float tsum = 0.f, tsq = 0.f;
#pragma unroll
    for (int w = 0; w < 8; ++w) { tsum += s1[w]; tsq += s2[w]; }
    float mu  = tsum * (1.0f / D_);
    float var = tsq * (1.0f / D_) - mu * mu;          // biased, like torch LN
    float rs  = rsqrtf(var + LN_EPS_);
    for (int i = threadIdx.x; i < D_; i += 256) {
        float v = (xr[i] - mu) * rs * g[i] + bt[i];
        xn[row * D_ + i] = (_Float16)v;
    }
}

// ---------------------------------------------------------------------------
// Kernel 2: convert w_in (3D x D) and w_out (D x D) f32 -> f16.
// ---------------------------------------------------------------------------
__global__ void cvt_kernel(const float* __restrict__ w_in,
                           const float* __restrict__ w_out,
                           _Float16* __restrict__ w_in_h,
                           _Float16* __restrict__ w_out_h) {
    const size_t n1 = (size_t)3 * D_ * D_;
    const size_t n2 = (size_t)D_ * D_;
    size_t stride = (size_t)gridDim.x * 256;
    for (size_t i = (size_t)blockIdx.x * 256 + threadIdx.x; i < n1 + n2; i += stride) {
        if (i < n1) w_in_h[i] = (_Float16)w_in[i];
        else        w_out_h[i - n1] = (_Float16)w_out[i - n1];
    }
}

// ---------------------------------------------------------------------------
// Kernel 3: QKV GEMM. qkv = xn @ w_in^T + b_in.
// M = B*S = 4096, N = 3*D = 3072, K = D = 1024. One 16x16 tile per wave.
// Scatters q,k into (B,H,S,HD) and v transposed into (B,H,HD,S), all f16.
// ---------------------------------------------------------------------------
__global__ void qkv_gemm_kernel(const _Float16* __restrict__ xn,
                                const _Float16* __restrict__ w,
                                const float* __restrict__ b_in,
                                _Float16* __restrict__ qf,
                                _Float16* __restrict__ kf,
                                _Float16* __restrict__ vtf) {
    const int NT = (3 * D_) / 16;                 // 192 tiles along N
    int wave = threadIdx.x >> 5, lane = threadIdx.x & 31;
    int t  = blockIdx.x * 8 + wave;
    int mt = t / NT, nt = t % NT;
    int m0 = mt * 16, n0 = nt * 16;

    v8f c = {};
    for (int kk = 0; kk < D_; kk += 32) {
        v16h a = load_a_frag(xn + (size_t)m0 * D_ + kk, D_);
        v16h b = load_b_frag(w  + (size_t)n0 * D_ + kk, D_);
        c = wmma_f16(a, b, c);
    }

    int r = lane & 15, sel = lane >> 4;
    int n = n0 + r;
    float bias = b_in[n];
    int which = n >> 10;           // 0:q 1:k 2:v
    int d  = n & (D_ - 1);
    int h  = d >> 6;
    int hd = d & (HD_ - 1);
#pragma unroll
    for (int i = 0; i < 8; ++i) {
        int m = m0 + i + 8 * sel;                 // global token index
        int b_ = m >> 10;
        int s_ = m & (S_ - 1);
        _Float16 hv = (_Float16)(c[i] + bias);
        if (which == 2) {
            vtf[(((size_t)b_ * H_ + h) * HD_ + hd) * S_ + s_] = hv;
        } else {
            size_t off = (((size_t)b_ * H_ + h) * S_ + s_) * HD_ + hd;
            if (which == 0) qf[off] = hv; else kf[off] = hv;
        }
    }
}

// ---------------------------------------------------------------------------
// Kernel 4: fused scores + causal mask + softmax + (P @ V).
// One block per (b, h, 16-query tile). Score strip kept in f16 LDS.
// Writes f32 softmax weights to d_out and attn (f16, (B,S,D)) to ws.
// ---------------------------------------------------------------------------
__global__ void attn_kernel(const _Float16* __restrict__ qf,
                            const _Float16* __restrict__ kf,
                            const _Float16* __restrict__ vtf,
                            float* __restrict__ wout,
                            _Float16* __restrict__ attn) {
    __shared__ _Float16 sc[16][S_ + 8];           // padded: 16 x 1032 f16

    int wave = threadIdx.x >> 5, lane = threadIdx.x & 31;
    int r = lane & 15, sel = lane >> 4;
    int qt = blockIdx.x & 63;
    int h  = (blockIdx.x >> 6) & 15;
    int b  = blockIdx.x >> 10;
    int q0 = qt * 16;

    const _Float16* qbase = qf  + (((size_t)b * H_ + h) * S_) * HD_;   // (S,HD)
    const _Float16* kbase = kf  + (((size_t)b * H_ + h) * S_) * HD_;   // (S,HD)
    const _Float16* vbase = vtf + (((size_t)b * H_ + h) * HD_) * S_;   // (HD,S)

    const float scale   = 0.125f;                 // 1/sqrt(64)
    const float NEG_INF = -__builtin_inff();

    // ---- Phase 1: scores = scale * Q K^T, causal mask, into LDS ----
    v16h a0 = load_a_frag(qbase + (size_t)q0 * HD_ +  0, HD_);
    v16h a1 = load_a_frag(qbase + (size_t)q0 * HD_ + 32, HD_);
    for (int t = wave; t < S_ / 16; t += 8) {
        int k0 = t * 16;
        v16h b0 = load_b_frag(kbase + (size_t)k0 * HD_ +  0, HD_);
        v16h b1 = load_b_frag(kbase + (size_t)k0 * HD_ + 32, HD_);
        v8f c = {};
        c = wmma_f16(a0, b0, c);
        c = wmma_f16(a1, b1, c);
#pragma unroll
        for (int i = 0; i < 8; ++i) {
            int m   = i + 8 * sel;
            int col = k0 + r;
            float v = c[i] * scale;
            if (col > q0 + m) v = NEG_INF;        // causal mask
            sc[m][col] = (_Float16)v;
        }
    }
    __syncthreads();

    // ---- Phase 2: per-row softmax (2 rows per wave), write f32 weights ----
    for (int rr = 0; rr < 2; ++rr) {
        int row = wave * 2 + rr;
        float mx = NEG_INF;
        for (int c0 = lane; c0 < S_; c0 += 32)
            mx = fmaxf(mx, (float)sc[row][c0]);
#pragma unroll
        for (int m = 16; m >= 1; m >>= 1) mx = fmaxf(mx, __shfl_xor(mx, m, 32));
        float sum = 0.f;
        for (int c0 = lane; c0 < S_; c0 += 32) {
            float e = __expf((float)sc[row][c0] - mx);
            sc[row][c0] = (_Float16)e;
            sum += e;
        }
#pragma unroll
        for (int m = 16; m >= 1; m >>= 1) sum += __shfl_xor(sum, m, 32);
        float inv = 1.0f / sum;
        float* wrow = wout + (((size_t)b * H_ + h) * S_ + (q0 + row)) * S_;
        for (int c0 = lane; c0 < S_; c0 += 32) {
            float p = (float)sc[row][c0] * inv;
            sc[row][c0] = (_Float16)p;            // f16 copy for P@V
            wrow[c0] = p;                         // required f32 output
        }
    }
    __syncthreads();

    // ---- Phase 3: attn = P @ V  (waves 0..3 each own 16 of HD=64 cols) ----
    if (wave < 4) {
        int n0 = wave * 16;
        v8f c = {};
        for (int kk = 0; kk < S_; kk += 32) {
            v16h a;
#pragma unroll
            for (int j = 0; j < 8; ++j) {
                a[j]     = sc[r][kk + sel * 8 + j];
                a[j + 8] = sc[r][kk + 16 + sel * 8 + j];
            }
            v16h bf = load_b_frag(vbase + (size_t)n0 * S_ + kk, S_);
            c = wmma_f16(a, bf, c);
        }
#pragma unroll
        for (int i = 0; i < 8; ++i) {
            int m  = i + 8 * sel;
            int hd = n0 + r;
            int s  = q0 + m;
            attn[((size_t)b * S_ + s) * D_ + h * HD_ + hd] = (_Float16)c[i];
        }
    }
}

// ---------------------------------------------------------------------------
// Kernel 5: out = x + attn @ w_out^T + b_out.  M=4096, N=1024, K=1024.
// ---------------------------------------------------------------------------
__global__ void outproj_kernel(const _Float16* __restrict__ attn,
                               const _Float16* __restrict__ w,
                               const float* __restrict__ bo,
                               const float* __restrict__ x,
                               float* __restrict__ out) {
    const int NT = D_ / 16;                       // 64
    int wave = threadIdx.x >> 5, lane = threadIdx.x & 31;
    int t  = blockIdx.x * 8 + wave;
    int mt = t / NT, nt = t % NT;
    int m0 = mt * 16, n0 = nt * 16;

    v8f c = {};
    for (int kk = 0; kk < D_; kk += 32) {
        v16h a = load_a_frag(attn + (size_t)m0 * D_ + kk, D_);
        v16h b = load_b_frag(w    + (size_t)n0 * D_ + kk, D_);
        c = wmma_f16(a, b, c);
    }

    int r = lane & 15, sel = lane >> 4;
    int n = n0 + r;
    float bias = bo[n];
#pragma unroll
    for (int i = 0; i < 8; ++i) {
        int m = m0 + i + 8 * sel;
        size_t off = (size_t)m * D_ + n;
        out[off] = c[i] + bias + x[off];          // residual
    }
}

// ---------------------------------------------------------------------------
// Launcher
// ---------------------------------------------------------------------------
extern "C" void kernel_launch(void* const* d_in, const int* in_sizes, int n_in,
                              void* d_out, int out_size, void* d_ws, size_t ws_size,
                              hipStream_t stream) {
    const float* x     = (const float*)d_in[0];
    const float* w_in  = (const float*)d_in[1];
    const float* b_in  = (const float*)d_in[2];
    const float* w_out = (const float*)d_in[3];
    const float* b_out = (const float*)d_in[4];
    const float* ln_w  = (const float*)d_in[5];
    const float* ln_b  = (const float*)d_in[6];
    // d_in[7] = attn_mask (bool causal) — causality is hardcoded.

    // Workspace layout (halves). Total 48 MB.
    _Float16* ws = (_Float16*)d_ws;
    _Float16* xn_h    = ws;                                   // 4096*1024
    _Float16* w_in_h  = xn_h    + (size_t)4096 * 1024;        // 3072*1024
    _Float16* w_out_h = w_in_h  + (size_t)3072 * 1024;        // 1024*1024
    _Float16* q_h     = w_out_h + (size_t)1024 * 1024;        // B*H*S*HD
    _Float16* k_h     = q_h     + (size_t)B_ * H_ * S_ * HD_;
    _Float16* vt_h    = k_h     + (size_t)B_ * H_ * S_ * HD_;
    _Float16* attn_h  = vt_h    + (size_t)B_ * H_ * S_ * HD_; // (B,S,D)

    float* out_x = (float*)d_out;                             // (B,S,D)
    float* out_w = out_x + (size_t)B_ * S_ * D_;              // (B,H,S,S)

    ln_kernel<<<B_ * S_, 256, 0, stream>>>(x, ln_w, ln_b, xn_h);
    cvt_kernel<<<4096, 256, 0, stream>>>(w_in, w_out, w_in_h, w_out_h);
    // QKV: 256 x 192 tiles, 8 waves/block -> 6144 blocks
    qkv_gemm_kernel<<<6144, 256, 0, stream>>>(xn_h, w_in_h, b_in, q_h, k_h, vt_h);
    // Attention: one block per (b,h,16-query tile) -> 4*16*64 = 4096 blocks
    attn_kernel<<<B_ * H_ * (S_ / 16), 256, 0, stream>>>(q_h, k_h, vt_h, out_w, attn_h);
    // Out-proj: 256 x 64 tiles, 8 waves/block -> 2048 blocks
    outproj_kernel<<<2048, 256, 0, stream>>>(attn_h, w_out_h, b_out, x, out_x);
}
